// IGMSA_6983616824033
// MI455X (gfx1250) — matrix operand order repaired and checked by reading, not compiled, not run on hardware
//
#include <hip/hip_runtime.h>
#include <hip/hip_bf16.h>
#include <math.h>

// ---------------------------------------------------------------------------
// IGMSA for MI455X (gfx1250): wave32, WMMA 16x16x32 bf16 everywhere,
// TDM (tensor_load_to_lds) staging of the v-tile in the output GEMM.
// Shapes: B=4, C=256, H=W=256, N=65536, HEADS=8, DIM_HEAD=32, hd=256.
// ---------------------------------------------------------------------------

#define NTOK   65536   // N = H*W
#define CDIM   256     // C == h*d
#define TN     64      // n-tile per block in big GEMMs
#define HEADS  8

typedef __bf16 bf16;
typedef __attribute__((ext_vector_type(16))) __bf16 v16bf;
typedef __attribute__((ext_vector_type(8)))  float  v8f;
typedef unsigned int u32x4 __attribute__((ext_vector_type(4)));
typedef int          i32x8 __attribute__((ext_vector_type(8)));
typedef int          i32x4 __attribute__((ext_vector_type(4)));

#if defined(__has_builtin)
# if __has_builtin(__builtin_amdgcn_tensor_load_to_lds)
#  define HAVE_TDM 1
# endif
#endif

union F16x16 { v16bf v; uint4 q[2]; };   // one WMMA A/B fragment (32B)
union F16x8  { bf16 h[8]; uint4 q; };    // 8 packed bf16 (16B)

#define WMMA_BF16(A, B, C) \
  __builtin_amdgcn_wmma_f32_16x16x32_bf16(false, (A), false, (B), (short)0, (C), false, false)

// ---------------------------------------------------------------------------
// Kernel 0: weight conversion f32->bf16, zero atomic accumulators
// ---------------------------------------------------------------------------
__global__ __launch_bounds__(256) void k_prep(
    const float* __restrict__ Wq, const float* __restrict__ Wk,
    const float* __restrict__ Wv, const float* __restrict__ Wp,
    bf16* __restrict__ Wqb, bf16* __restrict__ Wkb,
    bf16* __restrict__ Wvb, bf16* __restrict__ Wpb,
    float* __restrict__ nq, float* __restrict__ nk, float* __restrict__ G)
{
  int idx = blockIdx.x * 256 + threadIdx.x;
  if (idx < CDIM * CDIM) {
    Wqb[idx] = (bf16)Wq[idx];
    Wkb[idx] = (bf16)Wk[idx];
    Wvb[idx] = (bf16)Wv[idx];
    Wpb[idx] = (bf16)Wp[idx];
  }
  if (idx < 4 * CDIM) { nq[idx] = 0.f; nk[idx] = 0.f; }
  if (idx < 4 * HEADS * 32 * 32) G[idx] = 0.f;
}

// ---------------------------------------------------------------------------
// Kernel 1: QKV projection (WMMA), illumination gating, sum-of-squares.
//   q,k -> bf16 [b, o, n]   (n-contiguous, for Gram contraction over n)
//   v gated -> bf16 [b, n, o] (o-contiguous, for output contraction over o)
//   v ungated -> bf16 [b, o, n] (image layout for depthwise convs)
// ---------------------------------------------------------------------------
__global__ __launch_bounds__(256) void k_qkv(
    const float* __restrict__ x_in, const float* __restrict__ illu,
    const bf16* __restrict__ Wq, const bf16* __restrict__ Wk,
    const bf16* __restrict__ Wv,
    bf16* __restrict__ qo, bf16* __restrict__ ko,
    bf16* __restrict__ vno, bf16* __restrict__ vconv,
    float* __restrict__ nq, float* __restrict__ nk)
{
  __shared__ __align__(16) bf16 lx[TN * 264];   // [n][c], padded stride 264

  int blk = blockIdx.x;
  int b   = blk >> 10;
  int n0  = (blk & 1023) * TN;
  int t   = threadIdx.x;

  // Stage x tile: lx[j][c] = x_in[b, c, n0+j], converted to bf16.
  for (int i = 0; i < 64; ++i) {
    int c = i * 4 + (t >> 6);
    int j = t & 63;
    float xv = x_in[(((size_t)(b * CDIM + c)) << 16) + n0 + j];
    lx[j * 264 + c] = (bf16)xv;
  }
  __syncthreads();

  int wave = t >> 5, lane = t & 31;
  int lm = lane & 15, lh = lane >> 4;

  // 192 output tiles: 3 matrices x (4 n-tiles x 16 o-tiles)
  for (int tile = wave; tile < 192; tile += 8) {
    int mat = tile / 64;
    int rem = tile & 63;
    int nt  = rem >> 4;
    int ot  = rem & 15;
    const bf16* W = (mat == 0) ? Wq : (mat == 1) ? Wk : Wv;
    int o = ot * 16 + lm;
    __builtin_prefetch(W + (size_t)o * CDIM, 0, 3);   // global_prefetch_b8

    v8f acc = {0.f,0.f,0.f,0.f,0.f,0.f,0.f,0.f};
    const bf16* lrow = &lx[(nt * 16 + lm) * 264];
    for (int kk = 0; kk < 8; ++kk) {
      int cb = kk * 32;
      F16x16 a, bb;
      // A frag (x): lanes 0-15 K=[cb..cb+7]+[cb+16..23], lanes 16-31 shifted by 8
      a.q[0] = *(const uint4*)(lrow + cb + lh * 8);
      a.q[1] = *(const uint4*)(lrow + cb + 16 + lh * 8);
      // B frag (W^T): K=[cb..cb+7]+[cb+16..23] along W[o,:], lanes 16-31 shifted by 8
      const bf16* wp = W + (size_t)o * CDIM + cb + lh * 8;
      bb.q[0] = *(const uint4*)(wp);
      bb.q[1] = *(const uint4*)(wp + 16);
      acc = WMMA_BF16(a.v, bb.v, acc);
    }

    // C layout: row (= n) = r + 8*lh, col (= o) = lm
    int nbase = n0 + nt * 16 + lh * 8;                 // 8 consecutive n
    size_t on = (((size_t)(b * CDIM + o)) << 16) + nbase;

    if (mat < 2) {
      F16x8 pk;
      float ss = 0.f;
      #pragma unroll
      for (int r = 0; r < 8; ++r) { float v = acc[r]; ss += v * v; pk.h[r] = (bf16)v; }
      bf16* dst = (mat == 0 ? qo : ko);
      *(uint4*)(dst + on) = pk.q;
      ss += __shfl_xor(ss, 16, 32);
      if (lh == 0) atomicAdd((mat == 0 ? nq : nk) + b * CDIM + o, ss);
    } else {
      F16x8 pk;
      float4 il0 = *(const float4*)(illu + on);
      float4 il1 = *(const float4*)(illu + on + 4);
      float il[8] = {il0.x, il0.y, il0.z, il0.w, il1.x, il1.y, il1.z, il1.w};
      #pragma unroll
      for (int r = 0; r < 8; ++r) pk.h[r] = (bf16)acc[r];
      *(uint4*)(vconv + on) = pk.q;                    // ungated, [o,n]
      #pragma unroll
      for (int r = 0; r < 8; ++r) {
        float g = acc[r] * il[r];                      // illumination gating
        vno[((size_t)b << 24) + (size_t)(nbase + r) * CDIM + o] = (bf16)g;
      }
    }
  }
}

// ---------------------------------------------------------------------------
// Kernel 2: depthwise conv1 + exact GELU -> t (bf16, [b,c,n])
// ---------------------------------------------------------------------------
__global__ __launch_bounds__(256) void k_conv1(
    const bf16* __restrict__ vconv, const float* __restrict__ w1,
    bf16* __restrict__ tbuf)
{
  size_t idx = (size_t)blockIdx.x * 256 + threadIdx.x;   // B*C*N elements
  int n = (int)(idx & 65535);
  size_t bc = idx >> 16;
  int c = (int)(bc & 255);
  int y = n >> 8, x = n & 255;
  const bf16* src = vconv + (bc << 16);
  const float* w = w1 + c * 9;
  float s = 0.f;
  #pragma unroll
  for (int dy = -1; dy <= 1; ++dy) {
    int yy = y + dy;
    if (yy < 0 || yy > 255) continue;
    const bf16* row = src + (yy << 8);
    #pragma unroll
    for (int dx = -1; dx <= 1; ++dx) {
      int xx = x + dx;
      if (xx < 0 || xx > 255) continue;
      s += (float)row[xx] * w[(dy + 1) * 3 + (dx + 1)];
    }
  }
  float g = 0.5f * s * (1.f + erff(s * 0.70710678118654752f));  // exact GELU
  tbuf[idx] = (bf16)g;
}

// ---------------------------------------------------------------------------
// Kernel 3: Gram matrices G[b,h,32,32] = sum_n k[d,n]*q[e,n] (WMMA over N)
// ---------------------------------------------------------------------------
__global__ __launch_bounds__(256) void k_gram(
    const bf16* __restrict__ qo, const bf16* __restrict__ ko,
    float* __restrict__ G)
{
  int blk = blockIdx.x;                 // b*64 + h*8 + chunk
  int chunk = blk & 7;
  int h = (blk >> 3) & 7;
  int b = blk >> 6;
  int wave = threadIdx.x >> 5, lane = threadIdx.x & 31;
  int lm = lane & 15, lh = lane >> 4;

  size_t base = ((size_t)(b * CDIM + h * 32)) << 16;
  v8f a00 = {0,0,0,0,0,0,0,0}, a01 = a00, a10 = a00, a11 = a00;

  int n0w = chunk * 8192 + wave * 1024;
  for (int it = 0; it < 32; ++it) {
    int n0 = n0w + it * 32;
    F16x16 fa0, fa1, fb0, fb1;
    const bf16* kp0 = ko + base + ((size_t)lm << 16) + n0;        // d = lm
    const bf16* kp1 = ko + base + ((size_t)(16 + lm) << 16) + n0; // d = 16+lm
    fa0.q[0] = *(const uint4*)(kp0 + lh * 8);
    fa0.q[1] = *(const uint4*)(kp0 + 16 + lh * 8);
    fa1.q[0] = *(const uint4*)(kp1 + lh * 8);
    fa1.q[1] = *(const uint4*)(kp1 + 16 + lh * 8);
    const bf16* qp0 = qo + base + ((size_t)lm << 16) + n0 + lh * 8;
    const bf16* qp1 = qo + base + ((size_t)(16 + lm) << 16) + n0 + lh * 8;
    fb0.q[0] = *(const uint4*)(qp0); fb0.q[1] = *(const uint4*)(qp0 + 16);
    fb1.q[0] = *(const uint4*)(qp1); fb1.q[1] = *(const uint4*)(qp1 + 16);
    a00 = WMMA_BF16(fa0.v, fb0.v, a00);
    a01 = WMMA_BF16(fa0.v, fb1.v, a01);
    a10 = WMMA_BF16(fa1.v, fb0.v, a10);
    a11 = WMMA_BF16(fa1.v, fb1.v, a11);
  }

  float* Gp = G + (((b << 3) + h) << 10);
  int dbase = lh * 8;
  #pragma unroll
  for (int r = 0; r < 8; ++r) {
    int d = r + dbase;
    atomicAdd(Gp + ((d) << 5) + lm,           a00[r]);
    atomicAdd(Gp + ((d) << 5) + 16 + lm,      a01[r]);
    atomicAdd(Gp + ((d + 16) << 5) + lm,      a10[r]);
    atomicAdd(Gp + ((d + 16) << 5) + 16 + lm, a11[r]);
  }
}

// ---------------------------------------------------------------------------
// Kernel 4: norms + rescale + softmax; stores attn^T bf16 [b,h,e,d]
// ---------------------------------------------------------------------------
__global__ void k_softmax(
    const float* __restrict__ G, const float* __restrict__ nq,
    const float* __restrict__ nk, const float* __restrict__ rescale,
    bf16* __restrict__ attnT)
{
  int bh = blockIdx.x;
  int h = bh & 7, b = bh >> 3;
  int d = threadIdx.x;                       // 32 threads = 1 wave
  const float* Gp = G + (bh << 10) + (d << 5);
  float invk = 1.f / fmaxf(sqrtf(nk[b * CDIM + h * 32 + d]), 1e-12f);
  float rs = rescale[h];
  float row[32];
  float mx = -INFINITY;
  #pragma unroll
  for (int e = 0; e < 32; ++e) {
    float invq = 1.f / fmaxf(sqrtf(nq[b * CDIM + h * 32 + e]), 1e-12f);
    float v = Gp[e] * invk * invq * rs;
    row[e] = v;
    mx = fmaxf(mx, v);
  }
  float s = 0.f;
  #pragma unroll
  for (int e = 0; e < 32; ++e) { row[e] = expf(row[e] - mx); s += row[e]; }
  float inv = 1.f / s;
  bf16* ap = attnT + (bh << 10);
  #pragma unroll
  for (int e = 0; e < 32; ++e) ap[(e << 5) + d] = (bf16)(row[e] * inv);
}

// ---------------------------------------------------------------------------
// Kernel 5: M[b,c,h*32+e] = sum_d Wp[c,h*32+d] * attn[b,h,d,e]  (WMMA)
// ---------------------------------------------------------------------------
__global__ __launch_bounds__(256) void k_mw(
    const bf16* __restrict__ Wp, const bf16* __restrict__ attnT,
    bf16* __restrict__ Mw)
{
  int bh = blockIdx.x;
  int h = bh & 7, b = bh >> 3;
  int wave = threadIdx.x >> 5, lane = threadIdx.x & 31;
  int lm = lane & 15, lh = lane >> 4;

  for (int tile = wave; tile < 32; tile += 8) {
    int ct = tile >> 1, et = tile & 1;
    int c = ct * 16 + lm;
    F16x16 a, bb;
    const bf16* wp = Wp + (size_t)c * CDIM + h * 32;
    a.q[0] = *(const uint4*)(wp + lh * 8);
    a.q[1] = *(const uint4*)(wp + 16 + lh * 8);
    const bf16* ap = attnT + (bh << 10) + (et * 16 + lm) * 32 + lh * 8;
    bb.q[0] = *(const uint4*)(ap);
    bb.q[1] = *(const uint4*)(ap + 16);
    v8f acc = {0,0,0,0,0,0,0,0};
    acc = WMMA_BF16(a.v, bb.v, acc);
    bf16* mp = Mw + ((size_t)b << 16) + h * 32 + et * 16 + lm;
    #pragma unroll
    for (int r = 0; r < 8; ++r)
      mp[(size_t)(ct * 16 + r + lh * 8) * CDIM] = (bf16)acc[r];
  }
}

// ---------------------------------------------------------------------------
// Kernel 6: out[b,c,n] = sum_o M[b,c,o]*v[b,n,o] + bp[c] + dwconv2(t)[c,n]
// The v tile [n0..n0+63, 0..255] is one contiguous 32KB bf16 block -> TDM DMA
// into LDS (Tensor Data Mover), then all 8 waves build B-fragments from LDS.
// ---------------------------------------------------------------------------
__global__ __launch_bounds__(256) void k_out(
    const bf16* __restrict__ Mw, const bf16* __restrict__ vno,
    const bf16* __restrict__ tbuf, const float* __restrict__ bp,
    const float* __restrict__ w2, float* __restrict__ out)
{
  __shared__ __align__(16) bf16 lv[TN * CDIM];   // 32 KB, [n_local][o]

  int blk = blockIdx.x;
  int b  = blk >> 10;
  int n0 = (blk & 1023) * TN;
  int t = threadIdx.x;
  int wave = t >> 5, lane = t & 31;
  int lm = lane & 15, lh = lane >> 4;

  const bf16* vsrc = vno + ((size_t)b << 24) + (size_t)n0 * CDIM;

#if defined(HAVE_TDM)
  if (wave == 0) {
    unsigned long long gaddr = (unsigned long long)(uintptr_t)vsrc;
    unsigned int laddr = (unsigned int)(uintptr_t)&lv[0];   // LDS byte offset
    // D# group 0: count=1 | lds_addr | global_addr[56:0] | type=2
    u32x4 g0;
    g0[0] = 1u;
    g0[1] = laddr;
    g0[2] = (unsigned int)(gaddr & 0xffffffffu);
    g0[3] = (unsigned int)((gaddr >> 32) & 0x01ffffffu) | 0x80000000u;
    // D# group 1: mask=0, data_size=8B(3); dims in 8-byte units:
    //   tensor_dim0 = tile_dim0 = stride0 = 4096, tensor_dim1 = tile_dim1 = 1
    i32x8 g1;
    g1[0] = (int)(3u << 16);          // [17:16] data_size=3 (8B), no flags
    g1[1] = (int)(4096u << 16);       // [63:48] tensor_dim0[15:0]
    g1[2] = (int)(1u << 16);          // [95:80] tensor_dim1[15:0]=1
    g1[3] = (int)(4096u << 16);       // [127:112] tile_dim0
    g1[4] = 1;                        // [143:128] tile_dim1=1
    g1[5] = 4096;                     // [191:160] tensor_dim0_stride lo
    g1[6] = 0;
    g1[7] = 0;
    i32x4 z4 = {0, 0, 0, 0};          // groups 2/3 unused (<=2D tensor)
    i32x8 z8 = {0, 0, 0, 0, 0, 0, 0, 0};
    __builtin_amdgcn_tensor_load_to_lds(g0, g1, z4, z4, z8, 0);
    __builtin_amdgcn_s_wait_tensorcnt(0);
  }
#else
  // Fallback: cooperative 128-bit copy (32KB / 16B / 256 thr = 8 iters)
  for (int i = t; i < TN * CDIM / 8; i += 256)
    ((uint4*)lv)[i] = ((const uint4*)vsrc)[i];
#endif
  __syncthreads();

  for (int tile = wave; tile < 64; tile += 8) {
    int ct = tile >> 2, nt = tile & 3;
    int c0 = ct * 16;
    int n = n0 + nt * 16 + lm;

    v8f acc = {0,0,0,0,0,0,0,0};
    const bf16* mrow = Mw + ((size_t)b << 16) + (size_t)(c0 + lm) * CDIM;
    const bf16* vrow = lv + (size_t)(nt * 16 + lm) * CDIM;
    for (int kk = 0; kk < 8; ++kk) {
      F16x16 a, bb;
      a.q[0] = *(const uint4*)(mrow + kk * 32 + lh * 8);
      a.q[1] = *(const uint4*)(mrow + kk * 32 + 16 + lh * 8);
      const bf16* vp = vrow + kk * 32 + lh * 8;
      bb.q[0] = *(const uint4*)(vp);
      bb.q[1] = *(const uint4*)(vp + 16);
      acc = WMMA_BF16(a.v, bb.v, acc);
    }

    // epilogue: c = c0 + r + 8*lh, col n fixed per lane
    int y = n >> 8, x = n & 255;
    #pragma unroll
    for (int r = 0; r < 8; ++r) {
      int c = c0 + r + lh * 8;
      const bf16* tc = tbuf + (((size_t)(b * CDIM + c)) << 16);
      __builtin_prefetch(tc + (y << 8), 0, 0);
      const float* w = w2 + c * 9;
      float s = 0.f;
      #pragma unroll
      for (int dy = -1; dy <= 1; ++dy) {
        int yy = y + dy;
        if (yy < 0 || yy > 255) continue;
        const bf16* trow = tc + (yy << 8);
        #pragma unroll
        for (int dx = -1; dx <= 1; ++dx) {
          int xx = x + dx;
          if (xx < 0 || xx > 255) continue;
          s += (float)trow[xx] * w[(dy + 1) * 3 + (dx + 1)];
        }
      }
      out[(((size_t)(b * CDIM + c)) << 16) + n] = acc[r] + bp[c] + s;
    }
  }
}

// ---------------------------------------------------------------------------
extern "C" void kernel_launch(void* const* d_in, const int* in_sizes, int n_in,
                              void* d_out, int out_size, void* d_ws, size_t ws_size,
                              hipStream_t stream) {
  const float* x_in   = (const float*)d_in[0];
  const float* illu   = (const float*)d_in[1];
  const float* Wq     = (const float*)d_in[2];
  const float* Wk     = (const float*)d_in[3];
  const float* Wv     = (const float*)d_in[4];
  const float* rescal = (const float*)d_in[5];
  const float* Wp     = (const float*)d_in[6];
  const float* bp     = (const float*)d_in[7];
  const float* w1     = (const float*)d_in[8];
  const float* w2     = (const float*)d_in[9];
  float* out = (float*)d_out;

  char* ws = (char*)d_ws;
  const size_t SZ_W  = (size_t)CDIM * CDIM * 2;          // 128 KB each, bf16
  const size_t SZ_BIG = (size_t)4 * CDIM * NTOK * 2;     // 128 MB each, bf16

  size_t off = 0;
  bf16* Wqb   = (bf16*)(ws + off); off += SZ_W;
  bf16* Wkb   = (bf16*)(ws + off); off += SZ_W;
  bf16* Wvb   = (bf16*)(ws + off); off += SZ_W;
  bf16* Wpb   = (bf16*)(ws + off); off += SZ_W;
  bf16* qo    = (bf16*)(ws + off); off += SZ_BIG;        // [b,o,n]
  bf16* ko    = (bf16*)(ws + off); off += SZ_BIG;        // [b,o,n]
  bf16* vno   = (bf16*)(ws + off); off += SZ_BIG;        // [b,n,o] gated
  bf16* vconv = (bf16*)(ws + off); off += SZ_BIG;        // [b,o,n] ungated
  bf16* tbuf  = (bf16*)(ws + off); off += SZ_BIG;        // gelu(conv1)
  float* nq   = (float*)(ws + off); off += 4 * CDIM * 4;
  float* nk   = (float*)(ws + off); off += 4 * CDIM * 4;
  float* G    = (float*)(ws + off); off += (size_t)4 * HEADS * 32 * 32 * 4;
  bf16* attnT = (bf16*)(ws + off); off += (size_t)4 * HEADS * 32 * 32 * 2;
  bf16* Mw    = (bf16*)(ws + off); off += (size_t)4 * CDIM * CDIM * 2;
  (void)ws_size; (void)n_in; (void)in_sizes; (void)out_size;

  k_prep<<<256, 256, 0, stream>>>(Wq, Wk, Wv, Wp, Wqb, Wkb, Wvb, Wpb, nq, nk, G);
  k_qkv<<<4096, 256, 0, stream>>>(x_in, illu, Wqb, Wkb, Wvb, qo, ko, vno, vconv, nq, nk);
  k_conv1<<<262144, 256, 0, stream>>>(vconv, w1, tbuf);
  k_gram<<<256, 256, 0, stream>>>(qo, ko, G);
  k_softmax<<<32, 32, 0, stream>>>(G, nq, nk, rescal, attnT);
  k_mw<<<32, 256, 0, stream>>>(Wpb, attnT, Mw);
  k_out<<<4096, 256, 0, stream>>>(Mw, vno, tbuf, bp, w2, out);
}
